// MoELayer_14465449853190
// MI455X (gfx1250) — matrix operand order repaired and checked by reading, not compiled, not run on hardware
//
#include <hip/hip_runtime.h>
#include <hip/hip_bf16.h>
#include <math.h>

typedef __attribute__((ext_vector_type(16))) __bf16 v16bf;
typedef __attribute__((ext_vector_type(8)))  float  v8f;

namespace {

constexpr int Cdim  = 768;
constexpr int Ntok  = 4 * 2048;            // 8192 tokens
constexpr int NE    = 8;                   // experts
constexpr int NSLOT = Ntok * 2;            // top-2 -> 16384 routed slots

constexpr int BM = 128, BN = 64, BK = 32;  // block tile
constexpr int SLOT_CAP = NSLOT + NE * BM;  // per-expert BM-alignment padding
constexpr int ASTR = BK + 8;               // LDS pad vs bank conflicts
constexpr int BSTR = BK + 8;

union Frag {
  v16bf v;
  uint4 q[2];
};

__device__ __forceinline__ v8f wmma_bf16(const Frag& a, const Frag& b, v8f c) {
  return __builtin_amdgcn_wmma_f32_16x16x32_bf16(false, a.v, false, b.v,
                                                 (short)0, c, false, false);
}

// ---------------- router: one wave32 per token ----------------
__global__ void router_kern(const float* __restrict__ x,
                            const float* __restrict__ rw,
                            const float* __restrict__ bias,
                            int* __restrict__ idx, float* __restrict__ tw,
                            int* __restrict__ counts) {
  const int lane = threadIdx.x & 31;
  const int warp = threadIdx.x >> 5;
  const int tok  = blockIdx.x * 8 + warp;
  if (tok >= Ntok) return;

  float p[NE];
#pragma unroll
  for (int e = 0; e < NE; ++e) p[e] = 0.f;

  const float* xr = x + (size_t)tok * Cdim;
  for (int c = lane; c < Cdim; c += 32) {
    float xv = xr[c];
#pragma unroll
    for (int e = 0; e < NE; ++e) p[e] += xv * rw[e * Cdim + c];
  }
#pragma unroll
  for (int e = 0; e < NE; ++e) {
#pragma unroll
    for (int off = 16; off > 0; off >>= 1)
      p[e] += __shfl_xor(p[e], off, 32);
  }
  if (lane == 0) {
    float s[NE], b[NE];
#pragma unroll
    for (int e = 0; e < NE; ++e) {
      s[e] = 1.f / (1.f + __expf(-p[e]));
      b[e] = s[e] + bias[e];
    }
    int i0 = 0;
#pragma unroll
    for (int e = 1; e < NE; ++e) if (b[e] > b[i0]) i0 = e;
    int i1 = (i0 == 0) ? 1 : 0;
#pragma unroll
    for (int e = 0; e < NE; ++e) if (e != i0 && b[e] > b[i1]) i1 = e;
    float w0 = s[i0], w1 = s[i1];
    float dsum = w0 + w1 + 1e-20f;
    idx[tok * 2 + 0] = i0;
    idx[tok * 2 + 1] = i1;
    tw[tok * 2 + 0]  = w0 / dsum;
    tw[tok * 2 + 1]  = w1 / dsum;
    atomicAdd(&counts[i0], 1);
    atomicAdd(&counts[i1], 1);
  }
}

__global__ void zero_counts(int* counts) {
  if (threadIdx.x < NE) counts[threadIdx.x] = 0;
}

// BM-aligned exclusive scan of per-expert counts (+ zero cursors)
__global__ void scan_kern(const int* __restrict__ counts,
                          int* __restrict__ aoff, int* __restrict__ cur) {
  if (threadIdx.x == 0 && blockIdx.x == 0) {
    int acc = 0;
    for (int e = 0; e < NE; ++e) {
      aoff[e] = acc;
      acc += ((counts[e] + BM - 1) / BM) * BM;
      cur[e] = 0;
    }
    aoff[NE] = acc;
  }
}

__global__ void pad_init(int* __restrict__ tokrow, float* __restrict__ twp) {
  int i = blockIdx.x * blockDim.x + threadIdx.x;
  if (i < SLOT_CAP) { tokrow[i] = -1; twp[i] = 0.f; }
}

__global__ void fill_kern(const int* __restrict__ idx, const float* __restrict__ tw,
                          const int* __restrict__ aoff, int* __restrict__ cur,
                          int* __restrict__ tokrow, float* __restrict__ twp) {
  int n = blockIdx.x * blockDim.x + threadIdx.x;
  if (n >= Ntok) return;
#pragma unroll
  for (int k = 0; k < 2; ++k) {
    int e   = idx[n * 2 + k];
    int pos = atomicAdd(&cur[e], 1);
    int p   = aoff[e] + pos;
    tokrow[p] = n;
    twp[p]    = tw[n * 2 + k];
  }
}

// ---------------- WMMA GEMM (128x64 block tile, 32x32 wave tile) ----------------
// MODE 0: H_sh  = relu(x @ w_fc_sh)^2            (f32 A, bf16 out)
// MODE 1: out   = H_sh @ w_proj_sh               (bf16 A, f32 store)
// MODE 2: H_rt  = relu(gather(x) @ w1[e])^2      (gathered f32 A, bf16 out)
// MODE 3: out  += gate * (H_rt @ w2[e])          (bf16 A, atomic f32 accumulate)
template <int MODE>
__launch_bounds__(256)
__global__ void gemm_kern(const float* __restrict__ A32,
                          const __bf16* __restrict__ A16,
                          const float* __restrict__ Bsrc,
                          __bf16* __restrict__ Hout,
                          float* __restrict__ Fout,
                          const int* __restrict__ aoff,
                          const int* __restrict__ tokrow,
                          const float* __restrict__ twp) {
  __shared__ __bf16 As[BM][ASTR];
  __shared__ __bf16 Bst[BN][BSTR];   // transposed B tile: [n][k]

  const int tid = threadIdx.x;
  const int n0  = blockIdx.x * BN;

  int row0;
  const float* Bp = Bsrc;
  if (MODE == 2 || MODE == 3) {
    const int e = blockIdx.z;
    const int lo = aoff[e], hi = aoff[e + 1];
    row0 = lo + blockIdx.y * BM;
    if (row0 >= hi) return;                  // uniform per-block exit
    Bp = Bsrc + (size_t)e * Cdim * Cdim;
  } else {
    row0 = blockIdx.y * BM;
  }

  // A loader: 128x32 elems, each thread 2 rows x 8 k
  const int ar = tid >> 2;                   // 0..63 (also +64)
  const int ak = (tid & 3) << 3;             // 0,8,16,24
  // B loader: 32x64 elems, each thread 2 k-rows x 4 n (k-pair packing)
  const int bkr = (tid >> 4) << 1;           // 0,2,...,30
  const int bnn = (tid & 15) << 2;           // 0,4,...,60

  int arow0 = row0 + ar;
  int arow1 = row0 + ar + 64;
  if (MODE == 2) {
    int t0 = tokrow[arow0];
    int t1 = tokrow[arow1];
    arow0 = (t0 < 0) ? 0 : t0;               // pad rows read token 0
    arow1 = (t1 < 0) ? 0 : t1;
  }

  const int lane = tid & 31;
  const int wave = tid >> 5;
  const int wr = (wave & 3) << 5;            // wave row offset 0/32/64/96
  const int wc = (wave >> 2) << 5;           // wave col offset 0/32
  const int m = lane & 15;
  const int hsel = lane >> 4;

  v8f acc[2][2];
#pragma unroll
  for (int s = 0; s < 2; ++s)
#pragma unroll
    for (int t = 0; t < 2; ++t) acc[s][t] = (v8f){0, 0, 0, 0, 0, 0, 0, 0};

  for (int k0 = 0; k0 < Cdim; k0 += BK) {
    // ---- stage A tile ----
    if (MODE == 0 || MODE == 2) {            // fp32 A -> packed bf16, b128 store
#pragma unroll
      for (int h = 0; h < 2; ++h) {
        const int srow = (h == 0) ? arow0 : arow1;
        const float* src = A32 + (size_t)srow * Cdim + k0 + ak;
        float4 f0 = *(const float4*)(src);
        float4 f1 = *(const float4*)(src + 4);
        union { __bf16 hh[8]; uint4 q; } pk;
        pk.hh[0] = (__bf16)f0.x; pk.hh[1] = (__bf16)f0.y;
        pk.hh[2] = (__bf16)f0.z; pk.hh[3] = (__bf16)f0.w;
        pk.hh[4] = (__bf16)f1.x; pk.hh[5] = (__bf16)f1.y;
        pk.hh[6] = (__bf16)f1.z; pk.hh[7] = (__bf16)f1.w;
        *(uint4*)&As[ar + h * 64][ak] = pk.q;
      }
    } else {                                 // bf16 A straight copies (16B)
      *(uint4*)&As[ar][ak] =
          *(const uint4*)(A16 + (size_t)(row0 + ar) * Cdim + k0 + ak);
      *(uint4*)&As[ar + 64][ak] =
          *(const uint4*)(A16 + (size_t)(row0 + ar + 64) * Cdim + k0 + ak);
    }
    // ---- stage B tile transposed, packing (k,k+1) pairs -> b32 stores ----
    {
      const float* s0 = Bp + (size_t)(k0 + bkr) * Cdim + n0 + bnn;
      const float* s1 = s0 + Cdim;
      float4 f0 = *(const float4*)s0;
      float4 f1 = *(const float4*)s1;
      union { __bf16 hh[2]; unsigned u; } pk;
      pk.hh[0] = (__bf16)f0.x; pk.hh[1] = (__bf16)f1.x;
      *(unsigned*)&Bst[bnn + 0][bkr] = pk.u;
      pk.hh[0] = (__bf16)f0.y; pk.hh[1] = (__bf16)f1.y;
      *(unsigned*)&Bst[bnn + 1][bkr] = pk.u;
      pk.hh[0] = (__bf16)f0.z; pk.hh[1] = (__bf16)f1.z;
      *(unsigned*)&Bst[bnn + 2][bkr] = pk.u;
      pk.hh[0] = (__bf16)f0.w; pk.hh[1] = (__bf16)f1.w;
      *(unsigned*)&Bst[bnn + 3][bkr] = pk.u;
    }
    __syncthreads();

    // ISA 7.12.2 fragment packing (wave32):
    // A elem i -> k = ((i<8)? i : i+8) + hsel*8 ; row = m
    // B elem i -> k = ((i<8)? i : i+8) + hsel*8 ; col = m
    Frag fa0, fa1, fb0, fb1;
    fa0.q[0] = *(const uint4*)&As[wr + m][hsel * 8];
    fa0.q[1] = *(const uint4*)&As[wr + m][16 + hsel * 8];
    fa1.q[0] = *(const uint4*)&As[wr + 16 + m][hsel * 8];
    fa1.q[1] = *(const uint4*)&As[wr + 16 + m][16 + hsel * 8];
    fb0.q[0] = *(const uint4*)&Bst[wc + m][hsel * 8];
    fb0.q[1] = *(const uint4*)&Bst[wc + m][16 + hsel * 8];
    fb1.q[0] = *(const uint4*)&Bst[wc + 16 + m][hsel * 8];
    fb1.q[1] = *(const uint4*)&Bst[wc + 16 + m][16 + hsel * 8];

    acc[0][0] = wmma_bf16(fa0, fb0, acc[0][0]);
    acc[0][1] = wmma_bf16(fa0, fb1, acc[0][1]);
    acc[1][0] = wmma_bf16(fa1, fb0, acc[1][0]);
    acc[1][1] = wmma_bf16(fa1, fb1, acc[1][1]);
    __syncthreads();
  }

  // C/D layout: VGPR r -> row = r + hsel*8, col = lane&15
  const int colA = n0 + wc + m;
  const int colB = colA + 16;
#pragma unroll
  for (int s = 0; s < 2; ++s) {
#pragma unroll
    for (int r = 0; r < 8; ++r) {
      const int grow = row0 + wr + s * 16 + hsel * 8 + r;
      float v0 = acc[s][0][r], v1 = acc[s][1][r];
      if (MODE == 0 || MODE == 2) {
        v0 = v0 > 0.f ? v0 * v0 : 0.f;       // relu(x)^2
        v1 = v1 > 0.f ? v1 * v1 : 0.f;
        __bf16* o = Hout + (size_t)grow * Cdim;
        o[colA] = (__bf16)v0;
        o[colB] = (__bf16)v1;
      } else if (MODE == 1) {
        float* o = Fout + (size_t)grow * Cdim;
        o[colA] = v0;
        o[colB] = v1;
      } else {
        int t = tokrow[grow];
        if (t >= 0) {
          float w = twp[grow];
          float* o = Fout + (size_t)t * Cdim;
          atomicAdd(o + colA, w * v0);
          atomicAdd(o + colB, w * v1);
        }
      }
    }
  }
}

}  // namespace

extern "C" void kernel_launch(void* const* d_in, const int* in_sizes, int n_in,
                              void* d_out, int out_size, void* d_ws, size_t ws_size,
                              hipStream_t stream) {
  const float* x         = (const float*)d_in[0];
  const float* w_fc_sh   = (const float*)d_in[1];
  const float* w_proj_sh = (const float*)d_in[2];
  const float* w1        = (const float*)d_in[3];
  const float* w2        = (const float*)d_in[4];
  const float* router_w  = (const float*)d_in[5];
  const float* bias      = (const float*)d_in[6];
  float* out = (float*)d_out;
  (void)in_sizes; (void)n_in; (void)out_size; (void)ws_size;

  char* ws = (char*)d_ws;
  size_t off = 0;
  auto alloc = [&](size_t bytes) -> void* {
    void* p = ws + off;
    off = (off + bytes + 255) & ~(size_t)255;
    return p;
  };
  __bf16* H_sh = (__bf16*)alloc((size_t)Ntok * Cdim * sizeof(__bf16));
  __bf16* H_rt = (__bf16*)alloc((size_t)SLOT_CAP * Cdim * sizeof(__bf16));
  int*   idx    = (int*)alloc((size_t)Ntok * 2 * sizeof(int));
  float* tw     = (float*)alloc((size_t)Ntok * 2 * sizeof(float));
  int*   counts = (int*)alloc(64);
  int*   aoff   = (int*)alloc(64);
  int*   cur    = (int*)alloc(64);
  int*   tokrow = (int*)alloc((size_t)SLOT_CAP * sizeof(int));
  float* twp    = (float*)alloc((size_t)SLOT_CAP * sizeof(float));

  zero_counts<<<1, 32, 0, stream>>>(counts);
  router_kern<<<Ntok / 8, 256, 0, stream>>>(x, router_w, bias, idx, tw, counts);
  scan_kern<<<1, 1, 0, stream>>>(counts, aoff, cur);
  pad_init<<<(SLOT_CAP + 255) / 256, 256, 0, stream>>>(tokrow, twp);
  fill_kern<<<Ntok / 256, 256, 0, stream>>>(idx, tw, aoff, cur, tokrow, twp);

  dim3 blk(256);
  dim3 gsh(Cdim / BN, Ntok / BM, 1);       // (12, 64)
  dim3 grt(Cdim / BN, NSLOT / BM, NE);     // (12, 128, 8) worst-case; blocks exit early

  gemm_kern<0><<<gsh, blk, 0, stream>>>(x, nullptr, w_fc_sh, H_sh, nullptr,
                                        nullptr, nullptr, nullptr);
  gemm_kern<2><<<grt, blk, 0, stream>>>(x, nullptr, w1, H_rt, nullptr,
                                        aoff, tokrow, nullptr);
  gemm_kern<1><<<gsh, blk, 0, stream>>>(nullptr, H_sh, w_proj_sh, nullptr, out,
                                        nullptr, nullptr, nullptr);
  gemm_kern<3><<<grt, blk, 0, stream>>>(nullptr, H_rt, w2, nullptr, out,
                                        aoff, tokrow, twp);
}
